// GatedLinearAttention_50740743635384
// MI455X (gfx1250) — 8-bit matrix operand order repaired and checked by reading, compiled, not run on hardware
//
#include <hip/hip_runtime.h>
#include <hip/hip_bf16.h>
#include <cstdint>


typedef int v8i __attribute__((ext_vector_type(8)));

#define Mtok   4096
#define Dmod   1024
#define Dkv    1024
#define Dvv    2048
#define Rlow   16
#define HEADS  4
#define DKH    256
#define DVH    512
#define Bb     2
#define Ll     2048
#define EPSF   1e-8f

__global__ void wabs_reduce_kernel(const float* __restrict__ w, int n,
                                   float* __restrict__ out) {
  __shared__ float red[256];
  float s = 0.f;
  for (int i = blockIdx.x * blockDim.x + threadIdx.x; i < n;
       i += gridDim.x * blockDim.x)
    s += fabsf(w[i]);
  red[threadIdx.x] = s;
  __syncthreads();
  for (int st = 128; st > 0; st >>= 1) {
    if (threadIdx.x < st) red[threadIdx.x] += red[threadIdx.x + st];
    __syncthreads();
  }
  if (threadIdx.x == 0) atomicAdd(out, red[0]);
}

__global__ void wquant_kernel(const float* __restrict__ w,
                              const float* __restrict__ wsum, float numel_inv,
                              int8_t* __restrict__ wq, int n) {
  float mean = fmaxf(wsum[0] * numel_inv, 1e-5f);
  float s = 1.0f / mean;
  for (int i = blockIdx.x * blockDim.x + threadIdx.x; i < n;
       i += gridDim.x * blockDim.x) {
    float q = rintf(w[i] * s);
    q = fminf(fmaxf(q, -1.f), 1.f);
    wq[i] = (int8_t)(int)q;
  }
}

__global__ void act_quant_kernel(const float* __restrict__ x,
                                 const float* __restrict__ nw,
                                 int8_t* __restrict__ q,
                                 float* __restrict__ dq, int D) {
  __shared__ float rss[256];
  __shared__ float rmx[256];
  const int m = blockIdx.x;
  const float* xr = x + (size_t)m * D;
  float ss = 0.f, mx = 0.f;
  for (int i = threadIdx.x; i < D; i += 256) {
    float xv = xr[i];
    ss += xv * xv;
    mx = fmaxf(mx, fabsf(xv * nw[i]));
  }
  rss[threadIdx.x] = ss;
  rmx[threadIdx.x] = mx;
  __syncthreads();
  for (int st = 128; st > 0; st >>= 1) {
    if (threadIdx.x < st) {
      rss[threadIdx.x] += rss[threadIdx.x + st];
      rmx[threadIdx.x] = fmaxf(rmx[threadIdx.x], rmx[threadIdx.x + st]);
    }
    __syncthreads();
  }
  float r = rsqrtf(rss[0] / (float)D + EPSF);
  float amax = fmaxf(rmx[0] * r, 1e-5f);
  float scale = 127.0f / amax;
  if (threadIdx.x == 0) dq[m] = amax * (1.0f / 127.0f);
  for (int i = threadIdx.x; i < D; i += 256) {
    float v = rintf(xr[i] * r * nw[i] * scale);
    v = fminf(fmaxf(v, -128.f), 127.f);
    q[(size_t)m * D + i] = (int8_t)(int)v;
  }
}

template <int NT>
__global__ void gemm_iu8_kernel(const int8_t* __restrict__ A,
                                const float* __restrict__ adq,
                                const int8_t* __restrict__ Bw,
                                const float* __restrict__ wsum,
                                float wnumel_inv,
                                const float* __restrict__ bias,
                                float* __restrict__ C, int M, int N, int K) {
  __shared__ int8_t sB[2][NT * 16 * 64];

  const int lane = threadIdx.x & 31;
  const int wave = threadIdx.x >> 5;
  const int nblock = blockIdx.x * (NT * 16);
  const int mbase = (blockIdx.y * 8 + wave) * 16;

  const int8_t* Ap = A + (size_t)(mbase + (lane & 15)) * K;
  const int koffA = (lane >> 4) * 8;

  const int crow = threadIdx.x >> 2;
  const int cseg = (threadIdx.x & 3) << 4;
  const int8_t* gB = Bw + (size_t)(nblock + crow) * K + cseg;

  auto issueB = [&](int buf, int kb) {
    if (threadIdx.x < NT * 64) {
      const int8_t* gp = gB + kb;
      unsigned lofs = (unsigned)(uintptr_t)&sB[buf][crow * 64 + cseg];
      asm volatile("global_load_async_to_lds_b128 %0, %1, off"
                   :: "v"(lofs), "v"(gp) : "memory");
    }
  };

  v8i acc[NT];
#pragma unroll
  for (int t = 0; t < NT; ++t) acc[t] = (v8i){};

  const int T = K >> 6;
  issueB(0, 0);

  for (int it = 0; it < T; ++it) {
    const int kb = it << 6;
    if (it + 1 < T) {
      issueB((it + 1) & 1, kb + 64);
      asm volatile("s_wait_asynccnt 0x1" ::: "memory");
    } else {
      asm volatile("s_wait_asynccnt 0x0" ::: "memory");
    }
    __syncthreads();

    __builtin_prefetch(Ap + kb + 64, 0, 0);
    v8i a;
#pragma unroll
    for (int j = 0; j < 8; ++j) {
      int ka = kb + (koffA << 1) + ((j >> 2) << 5) + ((j & 3) << 2);
      a[j] = *(const int*)(Ap + ka);
    }
    const int buf = it & 1;
#pragma unroll
    for (int t = 0; t < NT; ++t) {
      const int8_t* bp =
          &sB[buf][(t * 16 + (lane & 15)) * 64 + ((lane >> 4) << 4)];
      int4 lo = *(const int4*)bp;
      int4 hi = *(const int4*)(bp + 32);
      v8i b;
      b[0] = lo.x; b[1] = lo.y; b[2] = lo.z; b[3] = lo.w;
      b[4] = hi.x; b[5] = hi.y; b[6] = hi.z; b[7] = hi.w;
      acc[t] = __builtin_amdgcn_wmma_i32_16x16x64_iu8(
           true, a,  true, b, acc[t],
           false,  false);
    }
    __syncthreads();
  }

  const float wdq = fmaxf(wsum[0] * wnumel_inv, 1e-5f);
#pragma unroll
  for (int t = 0; t < NT; ++t) {
    const int ncol = nblock + t * 16 + (lane & 15);
    const float bval = bias ? bias[ncol] : 0.0f;
#pragma unroll
    for (int r = 0; r < 8; ++r) {
      int mrow = mbase + r + ((lane >> 4) << 3);
      C[(size_t)mrow * N + ncol] =
          (float)acc[t][r] * adq[mrow] * wdq + bval;
    }
  }
}

__global__ void gk2_kernel(const float* __restrict__ u,
                           const float* __restrict__ ngk2,
                           const int8_t* __restrict__ wq2,
                           const float* __restrict__ wsum2,
                           const float* __restrict__ bgk2,
                           float* __restrict__ gk, int M) {
  __shared__ int8_t qu[16];
  __shared__ float inv_u;
  const int m = blockIdx.x;
  if (threadIdx.x == 0) {
    float ss = 0.f, mx = 0.f;
    float uv[16];
#pragma unroll
    for (int i = 0; i < 16; ++i) {
      uv[i] = u[(size_t)m * 16 + i];
      ss += uv[i] * uv[i];
      mx = fmaxf(mx, fabsf(uv[i] * ngk2[i]));
    }
    float r = rsqrtf(ss * (1.0f / 16.0f) + EPSF);
    float amax = fmaxf(mx * r, 1e-5f);
    float scale = 127.0f / amax;
    inv_u = amax * (1.0f / 127.0f);
#pragma unroll
    for (int i = 0; i < 16; ++i) {
      float v = rintf(uv[i] * r * ngk2[i] * scale);
      qu[i] = (int8_t)(int)fminf(fmaxf(v, -128.f), 127.f);
    }
  }
  __syncthreads();
  float wdq = fmaxf(wsum2[0] * (1.0f / 16384.0f), 1e-5f);
#pragma unroll
  for (int j = 0; j < 4; ++j) {
    int n = threadIdx.x + j * 256;
    int s = 0;
#pragma unroll
    for (int i = 0; i < 16; ++i) s += (int)qu[i] * (int)wq2[n * 16 + i];
    float y = (float)s * inv_u * wdq + bgk2[n];
    float ls = fminf(y, 0.f) - log1pf(expf(-fabsf(y)));
    gk[(size_t)m * 1024 + n] = ls * (1.0f / 16.0f);
  }
}

__global__ void gla_scan_kernel(const float* __restrict__ q,
                                const float* __restrict__ k,
                                const float* __restrict__ v,
                                const float* __restrict__ gk,
                                float* __restrict__ o) {
  __shared__ float sq[256], sk[256], se[256], sv[32];
  const int b = blockIdx.x >> 2;
  const int h = blockIdx.x & 3;
  const int vbase = blockIdx.y * 32;
  const int vloc = threadIdx.x >> 3;
  const int kseg = (threadIdx.x & 7) * 32;

  float S[32];
#pragma unroll
  for (int i = 0; i < 32; ++i) S[i] = 0.f;

  for (int t = 0; t < Ll; ++t) {
    const size_t bqk = ((size_t)(b * Ll + t)) * Dkv + h * DKH;
    sq[threadIdx.x] = q[bqk + threadIdx.x] * 0.0625f;
    sk[threadIdx.x] = k[bqk + threadIdx.x];
    se[threadIdx.x] = expf(gk[bqk + threadIdx.x]);
    if (threadIdx.x < 32)
      sv[threadIdx.x] =
          v[((size_t)(b * Ll + t)) * Dvv + h * DVH + vbase + threadIdx.x];
    __syncthreads();

    const float vv = sv[vloc];
    float acc = 0.f;
#pragma unroll
    for (int kk = 0; kk < 32; ++kk) {
      int ki = kseg + kk;
      S[kk] = se[ki] * S[kk] + sk[ki] * vv;
      acc += sq[ki] * S[kk];
    }
    acc += __shfl_xor(acc, 1);
    acc += __shfl_xor(acc, 2);
    acc += __shfl_xor(acc, 4);
    if ((threadIdx.x & 7) == 0)
      o[((size_t)(b * Ll + t)) * Dvv + h * DVH + vbase + vloc] = acc;
    __syncthreads();
  }
}

__global__ void silu_gate_kernel(const float* __restrict__ og,
                                 const float* __restrict__ g,
                                 float* __restrict__ out, int n) {
  for (int i = blockIdx.x * blockDim.x + threadIdx.x; i < n;
       i += gridDim.x * blockDim.x) {
    float gv = g[i];
    out[i] = og[i] * (gv / (1.0f + expf(-gv)));
  }
}

extern "C" void kernel_launch(void* const* d_in, const int* in_sizes, int n_in,
                              void* d_out, int out_size, void* d_ws,
                              size_t ws_size, hipStream_t stream) {
  const float* x    = (const float*)d_in[0];
  const float* wq   = (const float*)d_in[1];
  const float* nq   = (const float*)d_in[2];
  const float* wk   = (const float*)d_in[3];
  const float* nk   = (const float*)d_in[4];
  const float* wv   = (const float*)d_in[5];
  const float* nv   = (const float*)d_in[6];
  const float* wg   = (const float*)d_in[7];
  const float* ng   = (const float*)d_in[8];
  const float* wgk1 = (const float*)d_in[9];
  const float* ngk1 = (const float*)d_in[10];
  const float* wgk2 = (const float*)d_in[11];
  const float* bgk2 = (const float*)d_in[12];
  const float* ngk2 = (const float*)d_in[13];
  const float* wo   = (const float*)d_in[14];
  const float* no   = (const float*)d_in[15];
  float* out = (float*)d_out;

  char* ws = (char*)d_ws;
  size_t off = 0;
  auto alloc = [&](size_t bytes) {
    char* p = ws + off;
    off = (off + bytes + 255) & ~(size_t)255;
    return p;
  };
  float* wsum = (float*)alloc(8 * sizeof(float));
  int8_t* wq_q   = (int8_t*)alloc((size_t)Dkv * Dmod);
  int8_t* wk_q   = (int8_t*)alloc((size_t)Dkv * Dmod);
  int8_t* wv_q   = (int8_t*)alloc((size_t)Dvv * Dmod);
  int8_t* wg_q   = (int8_t*)alloc((size_t)Dvv * Dmod);
  int8_t* wgk1_q = (int8_t*)alloc((size_t)Rlow * Dmod);
  int8_t* wgk2_q = (int8_t*)alloc((size_t)Dkv * Rlow);
  int8_t* wo_q   = (int8_t*)alloc((size_t)Dmod * Dvv);
  int8_t* xq_q   = (int8_t*)alloc((size_t)Mtok * Dmod);
  int8_t* xq_k   = (int8_t*)alloc((size_t)Mtok * Dmod);
  int8_t* xq_v   = (int8_t*)alloc((size_t)Mtok * Dmod);
  int8_t* xq_g   = (int8_t*)alloc((size_t)Mtok * Dmod);
  int8_t* xq_gk  = (int8_t*)alloc((size_t)Mtok * Dmod);
  int8_t* oq_q   = (int8_t*)alloc((size_t)Mtok * Dvv);
  float* dq_q  = (float*)alloc(Mtok * sizeof(float));
  float* dq_k  = (float*)alloc(Mtok * sizeof(float));
  float* dq_v  = (float*)alloc(Mtok * sizeof(float));
  float* dq_g  = (float*)alloc(Mtok * sizeof(float));
  float* dq_gk = (float*)alloc(Mtok * sizeof(float));
  float* dq_o  = (float*)alloc(Mtok * sizeof(float));
  float* qf   = (float*)alloc((size_t)Mtok * Dkv * 4);
  float* kf   = (float*)alloc((size_t)Mtok * Dkv * 4);
  float* vf   = (float*)alloc((size_t)Mtok * Dvv * 4);
  float* gf   = (float*)alloc((size_t)Mtok * Dvv * 4);
  float* gk1f = (float*)alloc((size_t)Mtok * Rlow * 4);
  float* gkf  = (float*)alloc((size_t)Mtok * Dkv * 4);
  float* ogla = (float*)alloc((size_t)Mtok * Dvv * 4);
  float* ogat = (float*)alloc((size_t)Mtok * Dvv * 4);
  (void)ws_size; (void)in_sizes; (void)n_in; (void)out_size;

  hipMemsetAsync(wsum, 0, 8 * sizeof(float), stream);

  struct WSpec { const float* w; int n; int8_t* q; int slot; };
  const WSpec wspec[7] = {
      {wq,   Dkv * Dmod, wq_q,   0}, {wk,   Dkv * Dmod, wk_q,   1},
      {wv,   Dvv * Dmod, wv_q,   2}, {wg,   Dvv * Dmod, wg_q,   3},
      {wgk1, Rlow * Dmod, wgk1_q, 4}, {wgk2, Dkv * Rlow, wgk2_q, 5},
      {wo,   Dmod * Dvv, wo_q,   6}};
  for (int i = 0; i < 7; ++i)
    wabs_reduce_kernel<<<256, 256, 0, stream>>>(wspec[i].w, wspec[i].n,
                                                wsum + wspec[i].slot);
  for (int i = 0; i < 7; ++i)
    wquant_kernel<<<256, 256, 0, stream>>>(wspec[i].w, wsum + wspec[i].slot,
                                           1.0f / (float)wspec[i].n,
                                           wspec[i].q, wspec[i].n);

  act_quant_kernel<<<Mtok, 256, 0, stream>>>(x, nq,   xq_q,  dq_q,  Dmod);
  act_quant_kernel<<<Mtok, 256, 0, stream>>>(x, nk,   xq_k,  dq_k,  Dmod);
  act_quant_kernel<<<Mtok, 256, 0, stream>>>(x, nv,   xq_v,  dq_v,  Dmod);
  act_quant_kernel<<<Mtok, 256, 0, stream>>>(x, ng,   xq_g,  dq_g,  Dmod);
  act_quant_kernel<<<Mtok, 256, 0, stream>>>(x, ngk1, xq_gk, dq_gk, Dmod);

  const float inv_nw = 1.0f / (float)(Dkv * Dmod);
  const float inv_nv = 1.0f / (float)(Dvv * Dmod);
  gemm_iu8_kernel<4><<<dim3(Dkv / 64, Mtok / 128), 256, 0, stream>>>(
      xq_q, dq_q, wq_q, wsum + 0, inv_nw, nullptr, qf, Mtok, Dkv, Dmod);
  gemm_iu8_kernel<4><<<dim3(Dkv / 64, Mtok / 128), 256, 0, stream>>>(
      xq_k, dq_k, wk_q, wsum + 1, inv_nw, nullptr, kf, Mtok, Dkv, Dmod);
  gemm_iu8_kernel<4><<<dim3(Dvv / 64, Mtok / 128), 256, 0, stream>>>(
      xq_v, dq_v, wv_q, wsum + 2, inv_nv, nullptr, vf, Mtok, Dvv, Dmod);
  gemm_iu8_kernel<4><<<dim3(Dvv / 64, Mtok / 128), 256, 0, stream>>>(
      xq_g, dq_g, wg_q, wsum + 3, inv_nv, nullptr, gf, Mtok, Dvv, Dmod);
  gemm_iu8_kernel<1><<<dim3(Rlow / 16, Mtok / 128), 256, 0, stream>>>(
      xq_gk, dq_gk, wgk1_q, wsum + 4, 1.0f / (float)(Rlow * Dmod), nullptr,
      gk1f, Mtok, Rlow, Dmod);

  gk2_kernel<<<Mtok, 256, 0, stream>>>(gk1f, ngk2, wgk2_q, wsum + 5, bgk2,
                                       gkf, Mtok);

  gla_scan_kernel<<<dim3(Bb * HEADS, DVH / 32), 256, 0, stream>>>(qf, kf, vf,
                                                                  gkf, ogla);

  silu_gate_kernel<<<1024, 256, 0, stream>>>(ogla, gf, ogat, Mtok * Dvv);

  act_quant_kernel<<<Mtok, 256, 0, stream>>>(ogat, no, oq_q, dq_o, Dvv);
  gemm_iu8_kernel<4><<<dim3(Dmod / 64, Mtok / 128), 256, 0, stream>>>(
      oq_q, dq_o, wo_q, wsum + 6, 1.0f / (float)(Dmod * Dvv), nullptr, out,
      Mtok, Dmod, Dvv);
}
